// ResBlock_20633022890308
// MI455X (gfx1250) — matrix operand order repaired and checked by reading, not compiled, not run on hardware
//
#include <hip/hip_runtime.h>
#include <hip/hip_bf16.h>
#include <cstdint>
#include <cstddef>

// Problem constants (fixed by the reference)
#define NVOX 65536
#define CINC 64
#define CMIDC 256
#define COUTC 128
#define KOFF 27
#define EPSV 1e-5f

typedef __attribute__((ext_vector_type(16))) __bf16 v16bf;
typedef __attribute__((ext_vector_type(8)))  float  v8f;
typedef __attribute__((ext_vector_type(4)))  unsigned int u32x4;
typedef __attribute__((ext_vector_type(8)))  int i32x8;
typedef __attribute__((ext_vector_type(4)))  int i32x4;

union Frag {
    uint4  u[2];
    v16bf  v;
};

#ifndef __has_builtin
#define __has_builtin(x) 0
#endif
#if __has_builtin(__builtin_amdgcn_tensor_load_to_lds)
#define HAVE_TDM 1
#else
#define HAVE_TDM 0
#endif

// ---------------------------------------------------------------------------
// TDM: DMA a 2D bf16 tile (rows x rowElems, row stride = strideElems) from
// global into LDS, inserting pad_amount dwords after every pad_interval dwords
// so LDS rows land at the padded (bank-conflict-free) stride.
// D# bitfields per CDNA5 ISA (group0 128b, group1 256b).
// ---------------------------------------------------------------------------
__device__ __forceinline__ void tdm_load_2d_bf16(
    const void* gaddr, unsigned ldsByteAddr, int rowElems, int rows,
    int strideElems, int padIntervalCode, int padAmountCode)
{
#if HAVE_TDM
    unsigned long long ga = (unsigned long long)(uintptr_t)gaddr;
    u32x4 g0 = {};
    g0[0] = 1u;                                         // count=1 (valid user D#)
    g0[1] = ldsByteAddr;                                // lds_addr
    g0[2] = (unsigned)(ga & 0xFFFFFFFFu);               // global_addr[31:0]
    g0[3] = (unsigned)((ga >> 32) & 0x01FFFFFFu)        // global_addr[56:32]
          | (2u << 30);                                 // type=2 ("image")
    i32x8 g1 = {};
    g1[0] = (1 << 16)                                   // data_size=1 (2 bytes)
          | (1 << 20)                                   // pad_enable
          | (padIntervalCode << 22)                     // pad_interval
          | (padAmountCode << 25);                      // pad_amount
    g1[1] = (int)((rowElems & 0xFFFF) << 16);           // tensor_dim0[15:0] @63:48
    g1[2] = (int)(((unsigned)rowElems >> 16) & 0xFFFF)  // tensor_dim0[31:16]
          | ((rows & 0xFFFF) << 16);                    // tensor_dim1[15:0]
    g1[3] = (int)(((unsigned)rows >> 16) & 0xFFFF)      // tensor_dim1[31:16]
          | ((rowElems & 0xFFFF) << 16);                // tile_dim0 @127:112
    g1[4] = rows & 0xFFFF;                              // tile_dim1
    g1[5] = strideElems;                                // tensor_dim0_stride[31:0]
    g1[6] = 0;                                          // stride hi / dim1_stride
    g1[7] = 0;
    i32x4 z = {};
#if __clang_major__ >= 23
    i32x8 z8 = {};
    __builtin_amdgcn_tensor_load_to_lds(g0, g1, z, z, z8, 0);
#else
    __builtin_amdgcn_tensor_load_to_lds(g0, g1, z, z, 0);
#endif
#endif
}

__device__ __forceinline__ void wait_tensorcnt0()
{
#if __has_builtin(__builtin_amdgcn_s_wait_tensorcnt)
    __builtin_amdgcn_s_wait_tensorcnt(0);
#endif
}

// ---------------------------------------------------------------------------
// Gathered GEMM:  out[n, co] = bias[co] + sum_k sum_c A[src(n,k), c] * Wt[k, co, c]
// Wt is PRE-TRANSPOSED to [K, CO, C] so both LDS stages are pure b128 copies
// and B fragments read K-contiguous runs (WMMA 16-bit B layout).
//
// Block = 256 threads (8 waves). Block tile = 32 rows x 128 cols.
// Wave w: rows [(w>>2)*16,+16), cols [(w&3)*32,+32) -> two v8f accumulators.
// Per k: one TDM (or b128 copy-loop) stages the whole 128xC weight slab,
// gather stages 32xC activations, then C/32 dual-WMMA steps between ONE
// barrier pair. LDS rows padded to C+8 elems for conflict-free b128 reads.
// ---------------------------------------------------------------------------
template <int C, int CO, int K, bool HASIDX>
__global__ __launch_bounds__(256) void gemm_gather_bf16(
    const __bf16* __restrict__ A, const int* __restrict__ idx,
    const __bf16* __restrict__ Wt, const float* __restrict__ bias,
    float* __restrict__ out)
{
    constexpr int ROW = C + 8;                 // padded LDS row stride (elems)
    constexpr int NCH = C / 32;                // K-chunks per kernel offset
    // TDM pad codes: interval = C/2 dwords (32 -> code 4, 128 -> code 6),
    // amount = 4 dwords (code 3) == 8 bf16 pad -> stride C+8.
    constexpr int PADI = (C == 64) ? 4 : 6;
    constexpr int PADA = 3;

    __shared__ __align__(16) __bf16 ldsA[32 * ROW];    // activations tile
    __shared__ __align__(16) __bf16 ldsB[128 * ROW];   // weight slab
    __shared__ int ldsIdx[32 * K];                     // gather indices

    const int row0 = blockIdx.x * 32;
    const int col0 = blockIdx.y * 128;
    const int tid  = threadIdx.x;
    const int lane = tid & 31;
    const int wave = tid >> 5;
    const int mrow = (wave >> 2) * 16;   // 0 or 16
    const int colw = (wave & 3) * 32;    // 0,32,64,96

    if constexpr (HASIDX) {
        for (int i = tid; i < 32 * K; i += 256)
            ldsIdx[i] = idx[(size_t)row0 * K + i];
    }
    __syncthreads();

    v8f acc0 = {}, acc1 = {};

    for (int k = 0; k < K; ++k) {
        const __bf16* wslab = Wt + ((size_t)k * CO + col0) * C;

#if HAVE_TDM
        if (wave == 0)   // one DMA per block; EXEC-independent, TENSORcnt-tracked
            tdm_load_2d_bf16(wslab, (unsigned)(uintptr_t)&ldsB[0],
                             C, 128, C, PADI, PADA);
#else
        {   // manual slab staging: 128 rows x C bf16, all b128
            constexpr int BCH = (128 * C) >> 3;
            for (int ch = tid; ch < BCH; ch += 256) {
                int e = ch << 3;
                int n = e / C;
                int c = e - n * C;
                *reinterpret_cast<uint4*>(&ldsB[n * ROW + c]) =
                    *reinterpret_cast<const uint4*>(wslab + (size_t)n * C + c);
            }
            if (k + 1 < K)
                __builtin_prefetch(wslab + (size_t)CO * C, 0, 1);
        }
#endif

        // ---- gather A tile: 32 rows x C bf16, b128 loads (idx from LDS) ----
        constexpr int ACH = (32 * C) >> 3;
        for (int ch = tid; ch < ACH; ch += 256) {
            int e = ch << 3;
            int r = e / C;
            int c = e - r * C;
            int src;
            if constexpr (HASIDX)
                src = ldsIdx[r * K + k];
            else
                src = row0 + r;
            *reinterpret_cast<uint4*>(&ldsA[r * ROW + c]) =
                *reinterpret_cast<const uint4*>(A + (size_t)src * C + c);
        }

#if HAVE_TDM
        if (wave == 0) wait_tensorcnt0();
#endif
        __syncthreads();

        // ---- C/32 dual-WMMA steps, no barriers in between ----
#pragma unroll
        for (int cc = 0; cc < NCH; ++cc) {
            const int c0 = cc * 32;
            Frag a, b0, b1;
            // A frag: lane<16 -> M=lane, K=c0+{0..15};
            //         lane>=16 -> M=lane-16, K=c0+{16..31}
            const int m  = mrow + (lane & 15);
            const int kb = (lane >> 4) << 4;
            a.u[0] = *reinterpret_cast<const uint4*>(&ldsA[m * ROW + c0 + kb]);
            a.u[1] = *reinterpret_cast<const uint4*>(&ldsA[m * ROW + c0 + kb + 8]);
            // B frags: col n = colw + (lane&15) (+16 for 2nd tile);
            //          lanes 0-15 K=c0..c0+15, lanes 16-31 K=c0+16..c0+31
            const int kf = (lane >> 4) << 4;
            const uint4* pb0 = reinterpret_cast<const uint4*>(
                &ldsB[(colw + (lane & 15)) * ROW + c0 + kf]);
            const uint4* pb1 = reinterpret_cast<const uint4*>(
                &ldsB[(colw + 16 + (lane & 15)) * ROW + c0 + kf]);
            b0.u[0] = pb0[0]; b0.u[1] = pb0[1];
            b1.u[0] = pb1[0]; b1.u[1] = pb1[1];

            acc0 = __builtin_amdgcn_wmma_f32_16x16x32_bf16(
                false, a.v, false, b0.v, (short)0, acc0, false, false);
            acc1 = __builtin_amdgcn_wmma_f32_16x16x32_bf16(
                false, a.v, false, b1.v, (short)0, acc1, false, false);
        }
        __syncthreads();
    }

    // ---- epilogue: C/D layout (lane<16: M=r; lane>=16: M=8+r; N=lane&15) ----
    const int nn0   = col0 + colw + (lane & 15);
    const int nn1   = nn0 + 16;
    const int mbase = row0 + mrow + ((lane >> 4) << 3);
    const float bv0 = bias[nn0];
    const float bv1 = bias[nn1];
#pragma unroll
    for (int r = 0; r < 8; ++r) {
        out[(size_t)(mbase + r) * CO + nn0] = acc0[r] + bv0;
        out[(size_t)(mbase + r) * CO + nn1] = acc1[r] + bv1;
    }
}

// ---------------------------------------------------------------------------
// Helpers: conversions, BN reductions, epilogues
// ---------------------------------------------------------------------------
__global__ void zero_f32(float* p, int n) {
    int i = blockIdx.x * 256 + threadIdx.x;
    if (i < n) p[i] = 0.f;
}

__global__ void cvt_f32_bf16(const float* __restrict__ s, __bf16* __restrict__ d, int n) {
    int i = blockIdx.x * 256 + threadIdx.x;
    if (i < n) d[i] = (__bf16)s[i];
}

// W [K, C, CO] (f32) -> Wt [K, CO, C] (bf16): WMMA-ready B layout.
__global__ void cvt_w_transpose(const float* __restrict__ W, __bf16* __restrict__ Wt,
                                int K, int C, int CO) {
    int i = blockIdx.x * 256 + threadIdx.x;
    if (i < K * C * CO) {
        int c    = i % C;
        int rest = i / C;
        int n    = rest % CO;
        int k    = rest / CO;
        Wt[i] = (__bf16)W[((size_t)k * C + c) * CO + n];
    }
}

// Per-channel partial sum / sumsq over a 128-row slab, atomically merged.
__global__ void bn_stats(const float* __restrict__ x, float* __restrict__ sum,
                         float* __restrict__ sumsq, int C) {
    int c = threadIdx.x;
    size_t r0 = (size_t)blockIdx.x * 128;
    float s = 0.f, s2 = 0.f;
    for (int r = 0; r < 128; ++r) {
        float v = x[(r0 + r) * C + c];
        s += v; s2 += v * v;
    }
    atomicAdd(&sum[c], s);
    atomicAdd(&sumsq[c], s2);
}

__global__ void bn_finalize(const float* __restrict__ sum, const float* __restrict__ sumsq,
                            const float* __restrict__ gamma, const float* __restrict__ beta,
                            float* __restrict__ scale, float* __restrict__ shift,
                            int C, float invN) {
    int c = threadIdx.x;
    if (c < C) {
        float m  = sum[c] * invN;
        float v  = sumsq[c] * invN - m * m;
        float sc = gamma[c] * rsqrtf(v + EPSV);
        scale[c] = sc;
        shift[c] = beta[c] - m * sc;
    }
}

// mid_bf16 = bf16(relu(bn1(mid)))
__global__ void bn_relu_cvt(const float* __restrict__ x, const float* __restrict__ scale,
                            const float* __restrict__ shift, __bf16* __restrict__ y,
                            int C, size_t total) {
    size_t i = (size_t)blockIdx.x * 256 + threadIdx.x;
    if (i < total) {
        int c = (int)(i % C);
        float v = fmaxf(x[i] * scale[c] + shift[c], 0.f);
        y[i] = (__bf16)v;
    }
}

// out = relu(bn2(out) + identity)   (in place on d_out)
__global__ void bn_add_relu(float* __restrict__ out, const float* __restrict__ ident,
                            const float* __restrict__ scale, const float* __restrict__ shift,
                            int C, size_t total) {
    size_t i = (size_t)blockIdx.x * 256 + threadIdx.x;
    if (i < total) {
        int c = (int)(i % C);
        out[i] = fmaxf(out[i] * scale[c] + shift[c] + ident[i], 0.f);
    }
}

// ---------------------------------------------------------------------------
extern "C" void kernel_launch(void* const* d_in, const int* in_sizes, int n_in,
                              void* d_out, int out_size, void* d_ws, size_t ws_size,
                              hipStream_t stream) {
    const float* feats = (const float*)d_in[0];
    const int*   nbr   = (const int*)  d_in[1];
    const float* W1    = (const float*)d_in[2];
    const float* b1    = (const float*)d_in[3];
    const float* g1    = (const float*)d_in[4];
    const float* be1   = (const float*)d_in[5];
    const float* W2    = (const float*)d_in[6];
    const float* b2    = (const float*)d_in[7];
    const float* g2    = (const float*)d_in[8];
    const float* be2   = (const float*)d_in[9];
    const float* Wid   = (const float*)d_in[10];
    const float* bid   = (const float*)d_in[11];
    float* out = (float*)d_out;

    // ---- workspace carve-up (~139 MB) ----
    char* ws = (char*)d_ws;
    size_t off = 0;
    auto carve = [&](size_t bytes) -> void* {
        void* p = ws + off;
        off += (bytes + 255) & ~(size_t)255;
        return p;
    };
    __bf16* featsB = (__bf16*)carve((size_t)NVOX * CINC * 2);
    __bf16* w1B    = (__bf16*)carve((size_t)KOFF * CINC * CMIDC * 2);   // [K,CO,C]
    __bf16* w2B    = (__bf16*)carve((size_t)KOFF * CMIDC * COUTC * 2);  // [K,CO,C]
    __bf16* widB   = (__bf16*)carve((size_t)CINC * COUTC * 2);          // [CO,C] already
    float*  mid    = (float*) carve((size_t)NVOX * CMIDC * 4);
    __bf16* midB   = (__bf16*)carve((size_t)NVOX * CMIDC * 2);
    float*  ident  = (float*) carve((size_t)NVOX * COUTC * 4);
    float*  stats  = (float*) carve(1536 * 4);
    float *sum1 = stats,        *sq1 = stats + 256, *sc1 = stats + 512, *sh1 = stats + 768;
    float *sum2 = stats + 1024, *sq2 = stats + 1152, *sc2 = stats + 1280, *sh2 = stats + 1408;

    // ---- 0. zero BN accumulators (every call: deterministic) ----
    zero_f32<<<(1536 + 255) / 256, 256, 0, stream>>>(stats, 1536);

    // ---- 1. fp32 -> bf16 conversions (weights pre-transposed to [K,CO,C]) ----
    {
        int n = NVOX * CINC;
        cvt_f32_bf16<<<(n + 255) / 256, 256, 0, stream>>>(feats, featsB, n);
        n = KOFF * CINC * CMIDC;
        cvt_w_transpose<<<(n + 255) / 256, 256, 0, stream>>>(W1, w1B, KOFF, CINC, CMIDC);
        n = KOFF * CMIDC * COUTC;
        cvt_w_transpose<<<(n + 255) / 256, 256, 0, stream>>>(W2, w2B, KOFF, CMIDC, COUTC);
        n = CINC * COUTC;  // Wid is [COUT, CIN] == [n][c]: straight convert
        cvt_f32_bf16<<<(n + 255) / 256, 256, 0, stream>>>(Wid, widB, n);
    }

    // ---- 2. conv1: gathered GEMM [N,64] x [27,256,64]t -> mid f32 ----
    gemm_gather_bf16<CINC, CMIDC, KOFF, true>
        <<<dim3(NVOX / 32, CMIDC / 128), 256, 0, stream>>>(featsB, nbr, w1B, b1, mid);

    // ---- 3. BN1 stats + finalize + apply(ReLU, cvt to bf16) ----
    bn_stats<<<NVOX / 128, CMIDC, 0, stream>>>(mid, sum1, sq1, CMIDC);
    bn_finalize<<<1, CMIDC, 0, stream>>>(sum1, sq1, g1, be1, sc1, sh1, CMIDC, 1.f / NVOX);
    {
        size_t total = (size_t)NVOX * CMIDC;
        bn_relu_cvt<<<(unsigned)((total + 255) / 256), 256, 0, stream>>>(
            mid, sc1, sh1, midB, CMIDC, total);
    }

    // ---- 4. conv2: gathered GEMM [N,256] x [27,128,256]t -> d_out f32 (raw) ----
    gemm_gather_bf16<CMIDC, COUTC, KOFF, true>
        <<<dim3(NVOX / 32, COUTC / 128), 256, 0, stream>>>(midB, nbr, w2B, b2, out);

    // ---- 5. identity Linear: [N,64] x [128,64]t -> ident (no gather, K=1) ----
    gemm_gather_bf16<CINC, COUTC, 1, false>
        <<<dim3(NVOX / 32, COUTC / 128), 256, 0, stream>>>(featsB, nullptr, widB, bid, ident);

    // ---- 6. BN2 stats + finalize + fused residual-add + ReLU (in place) ----
    bn_stats<<<NVOX / 128, COUTC, 0, stream>>>(out, sum2, sq2, COUTC);
    bn_finalize<<<1, COUTC, 0, stream>>>(sum2, sq2, g2, be2, sc2, sh2, COUTC, 1.f / NVOX);
    {
        size_t total = (size_t)NVOX * COUTC;
        bn_add_relu<<<(unsigned)((total + 255) / 256), 256, 0, stream>>>(
            out, ident, sc2, sh2, COUTC, total);
    }
}
